// CrossAttentionModule_21440476742283
// MI455X (gfx1250) — matrix operand order repaired and checked by reading, not compiled, not run on hardware
//
#include <hip/hip_runtime.h>
#include <hip/hip_bf16.h>

// ---------------------------------------------------------------------------
// Cross-attention, MI455X (gfx1250), wave32 + v_wmma_f32_16x16x32_bf16.
//   proj:  q = Wq x1 + bq ; k = Wk x2 + bk ; v = Wv x2 + bv   (bf16 staging)
//   attn:  flash-attention, K/V tiles staged through LDS (shared by 4 waves),
//          next tile prefetched into registers while current tile computes.
// Layouts: qbuf,kbuf [B][N][C] bf16 ; vbuf [B][C][N] bf16 ; out [B][C][N] f32
// ---------------------------------------------------------------------------

#define BB 4
#define CC 256
#define NN 4096

#define KS 264   // LDS K-tile row stride (halves): 256 + 8 pad (bank spread)
#define VS 40    // LDS V-tile row stride (halves): 32 + 8 pad  (bank spread)

typedef __attribute__((ext_vector_type(16))) __bf16         v16bf;
typedef __attribute__((ext_vector_type(8)))  float          v8f;
typedef __attribute__((ext_vector_type(8)))  unsigned short v8u;
typedef __attribute__((ext_vector_type(16))) unsigned short v16u;

static __device__ __forceinline__ unsigned short f32_to_bf16_rne(float f) {
    union { float f; unsigned int u; } c; c.f = f;
    unsigned int u = c.u + 0x7FFFu + ((c.u >> 16) & 1u);
    return (unsigned short)(u >> 16);
}

static __device__ __forceinline__ v16bf as_bf16x16(v16u x) {
    union { v16u u; v16bf b; } c; c.u = x; return c.b;
}

static __device__ __forceinline__ v16bf frag16(const unsigned short* p) {
    // 16 contiguous halves via two 16B loads (no 32B-alignment assumption)
    v8u lo = *(const v8u*)(p);
    v8u hh = *(const v8u*)(p + 8);
    return as_bf16x16(__builtin_shufflevector(
        lo, hh, 0, 1, 2, 3, 4, 5, 6, 7, 8, 9, 10, 11, 12, 13, 14, 15));
}

// ---------------------------------------------------------------------------
// Kernel 1: fused q/k/v projections, f32 -> bf16 staging buffers.
// grid (NN, BB), block 256 (one thread per output channel).
// ---------------------------------------------------------------------------
__global__ __launch_bounds__(256) void ca_proj_kernel(
    const float* __restrict__ x1, const float* __restrict__ x2,
    const float* __restrict__ Wq, const float* __restrict__ bq,
    const float* __restrict__ Wk, const float* __restrict__ bk,
    const float* __restrict__ Wv, const float* __restrict__ bv,
    unsigned short* __restrict__ qbuf,   // [B][N][C]
    unsigned short* __restrict__ kbuf,   // [B][N][C]
    unsigned short* __restrict__ vbuf)   // [B][C][N]
{
    const int i = blockIdx.x;            // spatial index 0..N-1
    const int b = blockIdx.y;            // batch
    const int c = threadIdx.x;           // output channel

    __shared__ float xs1[CC];
    __shared__ float xs2[CC];
    xs1[c] = x1[((size_t)b * CC + c) * NN + i];
    xs2[c] = x2[((size_t)b * CC + c) * NN + i];
    __syncthreads();

    const float4* wq4 = (const float4*)(Wq + (size_t)c * CC);
    const float4* wk4 = (const float4*)(Wk + (size_t)c * CC);
    const float4* wv4 = (const float4*)(Wv + (size_t)c * CC);
    const float4* a1  = (const float4*)xs1;
    const float4* a2  = (const float4*)xs2;

    float q = bq[c], k = bk[c], v = bv[c];
#pragma unroll 8
    for (int t = 0; t < CC / 4; ++t) {
        float4 wq_ = wq4[t], wk_ = wk4[t], wv_ = wv4[t];
        float4 x1_ = a1[t], x2_ = a2[t];
        q += wq_.x * x1_.x + wq_.y * x1_.y + wq_.z * x1_.z + wq_.w * x1_.w;
        k += wk_.x * x2_.x + wk_.y * x2_.y + wk_.z * x2_.z + wk_.w * x2_.w;
        v += wv_.x * x2_.x + wv_.y * x2_.y + wv_.z * x2_.z + wv_.w * x2_.w;
    }

    qbuf[((size_t)b * NN + i) * CC + c] = f32_to_bf16_rne(q);
    kbuf[((size_t)b * NN + i) * CC + c] = f32_to_bf16_rne(k);
    vbuf[((size_t)b * CC + c) * NN + i] = f32_to_bf16_rne(v);
}

// ---------------------------------------------------------------------------
// Kernel 2: fused flash attention.
// 4 waves/block; each wave owns 16 query rows; K/V tiles (32 j-columns) are
// cooperatively staged in LDS, next tile prefetched into registers during
// compute. grid (NN/64, BB), block 128.
// ---------------------------------------------------------------------------
__global__ __launch_bounds__(128) void ca_attn_kernel(
    const unsigned short* __restrict__ qbuf,   // [B][N][C]
    const unsigned short* __restrict__ kbuf,   // [B][N][C]
    const unsigned short* __restrict__ vbuf,   // [B][C][N]
    float* __restrict__ out)                   // [B][C][N]
{
    const int b    = blockIdx.y;
    const int tid  = threadIdx.x;
    const int wave = tid >> 5;
    const int lane = tid & 31;
    const int hi   = lane >> 4;          // 0: lanes 0-15, 1: lanes 16-31
    const int l16  = lane & 15;
    const int i0   = blockIdx.x * 64 + wave * 16;   // query-row tile base

    __shared__ unsigned short ktile[32 * KS];       // K rows j..j+31, [j][c]
    __shared__ unsigned short vtile[CC * VS];       // V tile, [c][j]
    __shared__ unsigned short pst[4][16][32];       // per-wave P re-striping

    const unsigned short* kb_g = kbuf + (size_t)b * NN * CC;
    const unsigned short* vb_g = vbuf + (size_t)b * CC * NN;

    // ---- Q A-fragments (16x32 x 8 chunks covering C=256), kept in VGPRs ---
    // A layout: lane<16 holds row=lane, K {0..15};
    //           lane>=16 holds row=lane-16, K {16..31}.
    v16bf qf[8];
    {
        const unsigned short* qrow = qbuf + ((size_t)b * NN + (i0 + l16)) * CC;
#pragma unroll
        for (int cc = 0; cc < 8; ++cc) {
            const unsigned short* p = qrow + cc * 32 + hi * 16;
            v8u lo = *(const v8u*)(p);
            v8u hh = *(const v8u*)(p + 8);
            qf[cc] = as_bf16x16(__builtin_shufflevector(
                lo, hh, 0, 1, 2, 3, 4, 5, 6, 7, 8, 9, 10, 11, 12, 13, 14, 15));
        }
    }

    // ---- staging-register <-> tile mapping (per thread: 8 K + 8 V chunks) -
    // K: 32 rows x 256 halves = 1024 16B-chunks; V: 256 rows x 4 = 1024.
    int krow[8], kcol[8], vc[8], vp[8];
#pragma unroll
    for (int t = 0; t < 8; ++t) {
        int chk  = tid + t * 128;
        krow[t]  = chk >> 5;
        kcol[t]  = (chk & 31) * 8;
        vc[t]    = chk >> 2;
        vp[t]    = (chk & 3) * 8;
    }

    v8u kst[8], vst[8];
#pragma unroll
    for (int t = 0; t < 8; ++t) {   // prologue: stage tile j0 = 0
        kst[t] = *(const v8u*)(kb_g + (size_t)krow[t] * CC + kcol[t]);
        vst[t] = *(const v8u*)(vb_g + (size_t)vc[t] * NN + vp[t]);
    }

    // ---- online-softmax state: this lane's 8 rows (C/D striping) ----------
    float m_[8], l_[8];
#pragma unroll
    for (int r = 0; r < 8; ++r) { m_[r] = -1e30f; l_[r] = 0.0f; }
    v8f oacc[16];
#pragma unroll
    for (int n = 0; n < 16; ++n) oacc[n] = v8f{};

    for (int j0 = 0; j0 < NN; j0 += 32) {
        // ---- commit staged registers to LDS tiles -------------------------
#pragma unroll
        for (int t = 0; t < 8; ++t) {
            *(v8u*)(ktile + (size_t)krow[t] * KS + kcol[t]) = kst[t];
            *(v8u*)(vtile + (size_t)vc[t] * VS + vp[t])     = vst[t];
        }
        __syncthreads();

        // ---- prefetch next tile into registers (covered by compute) -------
        if (j0 + 32 < NN) {
            const unsigned short* kn = kb_g + (size_t)(j0 + 32) * CC;
            const unsigned short* vn = vb_g + (j0 + 32);
#pragma unroll
            for (int t = 0; t < 8; ++t) {
                kst[t] = *(const v8u*)(kn + (size_t)krow[t] * CC + kcol[t]);
                vst[t] = *(const v8u*)(vn + (size_t)vc[t] * NN + vp[t]);
            }
        }

        // ---- S tiles from LDS: s0 = Q K[:,j0..+15], s1 = Q K[:,j0+16..+31]
        v8f s0 = v8f{}, s1 = v8f{};
        const unsigned short* kb0l = ktile + (size_t)l16 * KS + hi * 16;
        const unsigned short* kb1l = kb0l + (size_t)16 * KS;
#pragma unroll
        for (int cc = 0; cc < 8; ++cc) {
            v16bf kf0 = frag16(kb0l + cc * 32);
            s0 = __builtin_amdgcn_wmma_f32_16x16x32_bf16(
                false, qf[cc], false, kf0, (short)0, s0, false, false);
            v16bf kf1 = frag16(kb1l + cc * 32);
            s1 = __builtin_amdgcn_wmma_f32_16x16x32_bf16(
                false, qf[cc], false, kf1, (short)0, s1, false, false);
        }

        // ---- online softmax (row reductions within each 16-lane half) -----
        float corr[8];
#pragma unroll
        for (int r = 0; r < 8; ++r) {
            float mx = fmaxf(s0[r], s1[r]);
#pragma unroll
            for (int d = 8; d >= 1; d >>= 1) mx = fmaxf(mx, __shfl_xor(mx, d, 32));
            float mn = fmaxf(m_[r], mx);
            corr[r]  = __expf(m_[r] - mn);
            float p0 = __expf(s0[r] - mn);
            float p1 = __expf(s1[r] - mn);
            s0[r] = p0;
            s1[r] = p1;
            float sum = p0 + p1;
#pragma unroll
            for (int d = 8; d >= 1; d >>= 1) sum += __shfl_xor(sum, d, 32);
            l_[r] = l_[r] * corr[r] + sum;
            m_[r] = mn;
        }

        // ---- rescale O accumulators by the max-correction -----------------
#pragma unroll
        for (int n = 0; n < 16; ++n)
#pragma unroll
            for (int r = 0; r < 8; ++r) oacc[n][r] *= corr[r];

        // ---- P (f32, C/D layout) -> bf16 LDS tile -> A-fragment -----------
#pragma unroll
        for (int r = 0; r < 8; ++r) {
            pst[wave][hi * 8 + r][l16]      = f32_to_bf16_rne(s0[r]);
            pst[wave][hi * 8 + r][16 + l16] = f32_to_bf16_rne(s1[r]);
        }
        asm volatile("s_wait_dscnt 0" ::: "memory");   // wave-local LDS RAW
        v16bf pf;
        {
            const unsigned short* pr = &pst[wave][l16][0] + hi * 16;
            v8u lo = *(const v8u*)(pr);
            v8u hh = *(const v8u*)(pr + 8);
            pf = as_bf16x16(__builtin_shufflevector(
                lo, hh, 0, 1, 2, 3, 4, 5, 6, 7, 8, 9, 10, 11, 12, 13, 14, 15));
        }

        // ---- O += P @ V, V B-fragments from LDS (K=j contiguous) ----------
        const unsigned short* vl = vtile + (size_t)l16 * VS + hi * 16;
#pragma unroll
        for (int n = 0; n < 16; ++n) {
            v16bf vf = frag16(vl + (size_t)n * 16 * VS);
            oacc[n] = __builtin_amdgcn_wmma_f32_16x16x32_bf16(
                false, pf, false, vf, (short)0, oacc[n], false, false);
        }

        __syncthreads();   // all waves done reading tiles before next commit
    }

    // ---- normalize and store: out[b][c][i], i contiguous per lane ---------
    float* ob = out + (size_t)b * CC * NN;
#pragma unroll
    for (int n = 0; n < 16; ++n) {
        int c = n * 16 + l16;
        v8f res;
#pragma unroll
        for (int r = 0; r < 8; ++r) res[r] = oacc[n][r] / l_[r];
        *(v8f*)(ob + (size_t)c * NN + i0 + hi * 8) = res;
    }
}

// ---------------------------------------------------------------------------
extern "C" void kernel_launch(void* const* d_in, const int* in_sizes, int n_in,
                              void* d_out, int out_size, void* d_ws, size_t ws_size,
                              hipStream_t stream) {
    const float* x1 = (const float*)d_in[0];
    const float* x2 = (const float*)d_in[1];
    const float* Wq = (const float*)d_in[2];
    const float* bq = (const float*)d_in[3];
    const float* Wk = (const float*)d_in[4];
    const float* bk = (const float*)d_in[5];
    const float* Wv = (const float*)d_in[6];
    const float* bv = (const float*)d_in[7];
    float* out = (float*)d_out;

    // workspace: q,k [B][N][C] bf16 ; v [B][C][N] bf16  (3 * 8 MiB = 24 MiB)
    unsigned short* qbuf = (unsigned short*)d_ws;
    unsigned short* kbuf = qbuf + (size_t)BB * NN * CC;
    unsigned short* vbuf = kbuf + (size_t)BB * NN * CC;

    ca_proj_kernel<<<dim3(NN, BB), 256, 0, stream>>>(
        x1, x2, Wq, bq, Wk, bk, Wv, bv, qbuf, kbuf, vbuf);

    ca_attn_kernel<<<dim3(NN / 64, BB), 128, 0, stream>>>(
        qbuf, kbuf, vbuf, out);
}
